// GatedGNN_59133109731330
// MI455X (gfx1250) — matrix operand order repaired and checked by reading, not compiled, not run on hardware
//
#include <hip/hip_runtime.h>
#include <hip/hip_bf16.h>

#define Bb 32
#define Nn 1024
#define Ff 256
#define Hh 128
#define Ll 4
#define BN (Bb * Nn)

typedef __attribute__((ext_vector_type(16))) __bf16 v16bf;
typedef __attribute__((ext_vector_type(8)))  float  v8f;
typedef unsigned int u32x4 __attribute__((ext_vector_type(4)));
typedef int          i32x8 __attribute__((ext_vector_type(8)));
typedef int          i32x4 __attribute__((ext_vector_type(4)));

// ---------------------------------------------------------------------------
// WMMA fragment loaders matching the CDNA5 16x16x32 bf16 VGPR layouts.
// A (16x32, MxK): lane L holds row M=L%16; elems 0..7 = K[k0+8g .. k0+8g+7],
//                 elems 8..15 = K[k0+16+8g ..], g = L/16. Two 16B loads.
// B (32x16, KxN): lane L holds col N=L%16; same K order as A (8g / 16+8g).
//                 Loaded from a TRANSPOSED operand BT[N][K]: two 16B loads.
// D (16x16 f32): elem v -> (M = 8*(L/16)+v, N = L%16).
// ---------------------------------------------------------------------------

static __device__ __forceinline__ v16bf
frag_a_bf16(const __bf16* __restrict__ A, int row0, int k0, int ld, int lane) {
  const __bf16* p = A + (size_t)(row0 + (lane & 15)) * ld + k0 + 8 * (lane >> 4);
  v16bf r;
  ((float4*)&r)[0] = *(const float4*)p;          // K = k0+8g .. +7
  ((float4*)&r)[1] = *(const float4*)(p + 16);   // K = k0+16+8g .. +7
  return r;
}

// A-fragment from an LDS tile laid out row-major [16+][64] bf16.
static __device__ __forceinline__ v16bf
frag_a_lds(const __bf16* base, int k0, int lane) {
  const __bf16* p = base + (lane & 15) * 64 + k0 + 8 * (lane >> 4);
  v16bf r;
  ((float4*)&r)[0] = *(const float4*)p;          // ds_load_b128
  ((float4*)&r)[1] = *(const float4*)(p + 16);
  return r;
}

static __device__ __forceinline__ v16bf
frag_a_f32(const float* __restrict__ A, int row0, int k0, int ld, int lane) {
  const float* p = A + (size_t)(row0 + (lane & 15)) * ld + k0 + 8 * (lane >> 4);
  float lo[8], hi[8];
  *(float4*)(lo + 0) = *(const float4*)(p + 0);
  *(float4*)(lo + 4) = *(const float4*)(p + 4);
  *(float4*)(hi + 0) = *(const float4*)(p + 16);
  *(float4*)(hi + 4) = *(const float4*)(p + 20);
  v16bf r;
  __bf16* e = (__bf16*)&r;
#pragma unroll
  for (int i = 0; i < 8; ++i) { e[i] = (__bf16)lo[i]; e[8 + i] = (__bf16)hi[i]; }
  return r;
}

static __device__ __forceinline__ v16bf
frag_b_fromT(const __bf16* __restrict__ BT, int n0, int k0, int ld, int lane) {
  const __bf16* p = BT + (size_t)(n0 + (lane & 15)) * ld + k0 + 8 * (lane >> 4);
  v16bf r;
  ((float4*)&r)[0] = *(const float4*)p;
  ((float4*)&r)[1] = *(const float4*)(p + 16);
  return r;
}

static __device__ __forceinline__ void
load_b_panel(v16bf* b, const __bf16* __restrict__ BT, int k0, int ld, int lane) {
#pragma unroll
  for (int j = 0; j < 8; ++j) b[j] = frag_b_fromT(BT, 16 * j, k0, ld, lane);
}

static __device__ __forceinline__ void
wmma8(v8f* acc, v16bf af, const v16bf* b) {
#pragma unroll
  for (int j = 0; j < 8; ++j)
    acc[j] = __builtin_amdgcn_wmma_f32_16x16x32_bf16(false, af, false, b[j],
                                                     (short)0, acc[j], false, false);
}

static __device__ __forceinline__ void
wmma_step(v8f* acc, v16bf af, const __bf16* __restrict__ BT, int k0, int ld, int lane) {
  v16bf b[8];
  load_b_panel(b, BT, k0, ld, lane);
  wmma8(acc, af, b);
}

// ---------------------------------------------------------------------------
// TDM: DMA a 2-D bf16 tile (tile_dim0=64 K-elems x tile_dim1=128 rows) from a
// row-major [1024 x 1024] bf16 tensor into LDS. D# per ISA 08_async_tensor §8.
// This toolchain exposes the 6-arg builtin (extra int32x8 group, then cpol).
// ---------------------------------------------------------------------------
static __device__ __forceinline__ void
tdm_load_tile_64x128(const __bf16* gsrc, __bf16* lds_dst) {
  unsigned long long ga = (unsigned long long)(uintptr_t)gsrc;
  unsigned lds = (unsigned)(uintptr_t)lds_dst;       // low 32 bits = LDS offset
  u32x4 g0;
  g0[0] = 1u;                                        // count=1, no gather/restore
  g0[1] = lds;                                       // lds_addr [63:32]
  g0[2] = (unsigned)ga;                              // global_addr [95:64]
  g0[3] = (unsigned)((ga >> 32) & 0x01FFFFFFu)       // global_addr [120:96]
          | 0x80000000u;                             // type=2 ("image") [127:126]
  i32x8 g1;
  g1[0] = 1 << 16;                                   // data_size=1 (2B/elem)
  g1[1] = (Nn & 0xFFFF) << 16;                       // tensor_dim0[15:0]
  g1[2] = ((Nn >> 16) & 0xFFFF) | ((Nn & 0xFFFF) << 16);   // td0 hi | td1 lo
  g1[3] = ((Nn >> 16) & 0xFFFF) | (64 << 16);        // td1 hi | tile_dim0=64
  g1[4] = 128;                                       // tile_dim1=128 (tile_dim2=0)
  g1[5] = Nn;                                        // tensor_dim0_stride[31:0]
  g1[6] = 0;                                         // stride hi | dim1_stride lo
  g1[7] = 0;
  i32x4 gz4 = {0, 0, 0, 0};                          // groups 2/3 unused (2-D)
  i32x8 gz8 = {0, 0, 0, 0, 0, 0, 0, 0};
  __builtin_amdgcn_tensor_load_to_lds(g0, g1, gz4, gz4, gz8, 0);
}

// ---------------------------------------------------------------------------
// prep: transposed bf16 weights ([Wz0;Wz1]^T etc. as [Hout][2H]), fused biases
// ---------------------------------------------------------------------------
__global__ void prep_kernel(const float* __restrict__ We,
                            const float* __restrict__ Wz0, const float* __restrict__ Wz1,
                            const float* __restrict__ Wr0, const float* __restrict__ Wr1,
                            const float* __restrict__ Wh0, const float* __restrict__ Wh1,
                            const float* __restrict__ bz0, const float* __restrict__ bz1,
                            const float* __restrict__ br0, const float* __restrict__ br1,
                            const float* __restrict__ bh0, const float* __restrict__ bh1,
                            __bf16* __restrict__ WeT, __bf16* __restrict__ WzT,
                            __bf16* __restrict__ WrT, __bf16* __restrict__ WhT,
                            float* __restrict__ bz, float* __restrict__ br,
                            float* __restrict__ bh) {
  int t = blockIdx.x * blockDim.x + threadIdx.x;
  if (t < Hh * Ff) {                         // WeT[n][k] = We[k][n]
    int n = t / Ff, k = t % Ff;
    WeT[t] = (__bf16)We[(size_t)k * Hh + n];
  }
  if (t < Hh * 2 * Hh) {                     // W*T[n][k], k<H from W*0, k>=H from W*1
    int n = t / (2 * Hh), k = t % (2 * Hh);
    float z = (k < Hh) ? Wz0[(size_t)k * Hh + n] : Wz1[(size_t)(k - Hh) * Hh + n];
    float r = (k < Hh) ? Wr0[(size_t)k * Hh + n] : Wr1[(size_t)(k - Hh) * Hh + n];
    float h = (k < Hh) ? Wh0[(size_t)k * Hh + n] : Wh1[(size_t)(k - Hh) * Hh + n];
    WzT[t] = (__bf16)z; WrT[t] = (__bf16)r; WhT[t] = (__bf16)h;
  }
  if (t < Hh) { bz[t] = bz0[t] + bz1[t]; br[t] = br0[t] + br1[t]; bh[t] = bh0[t] + bh1[t]; }
}

// ---------------------------------------------------------------------------
// adj fp32 -> bf16 once (64 MB result is L2-resident for layers 2..4)
// ---------------------------------------------------------------------------
__global__ __launch_bounds__(256) void adjcvt_kernel(const float* __restrict__ adj,
                                                     __bf16* __restrict__ adjb, long nq) {
  long i = (long)blockIdx.x * blockDim.x + threadIdx.x;
  long stride = (long)gridDim.x * blockDim.x;
  for (; i < nq; i += stride) {
    __builtin_prefetch(adj + 4 * (i + stride), 0, 1);     // global_prefetch_b8
    float4 f = ((const float4*)adj)[i];
    __bf16 o[4] __attribute__((aligned(8))) = {(__bf16)f.x, (__bf16)f.y,
                                               (__bf16)f.z, (__bf16)f.w};
    ((double*)adjb)[i] = *(const double*)o;
  }
}

// ---------------------------------------------------------------------------
// embed: x = (inputs @ We + be) * mask   [BN,256]@[256,128]
// writes x fp32 (d_out), x bf16 (row-major), x^T bf16 ([b][H][node])
// ---------------------------------------------------------------------------
__global__ __launch_bounds__(256) void embed_kernel(const float* __restrict__ in,
                                                    const float* __restrict__ mask,
                                                    const __bf16* __restrict__ WeT,
                                                    const float* __restrict__ be,
                                                    float* __restrict__ x,
                                                    __bf16* __restrict__ xb,
                                                    __bf16* __restrict__ xT) {
  const int lane = threadIdx.x & 31, wave = threadIdx.x >> 5;
  const int row0 = blockIdx.x * 128 + wave * 16;
  const v8f vzero = {0.f, 0.f, 0.f, 0.f, 0.f, 0.f, 0.f, 0.f};
  v8f acc[8];
#pragma unroll
  for (int j = 0; j < 8; ++j) acc[j] = vzero;

  v16bf a_cur = frag_a_f32(in, row0, 0, Ff, lane);
  v16bf b_cur[8];
  load_b_panel(b_cur, WeT, 0, Ff, lane);
#pragma unroll
  for (int ks = 0; ks < Ff; ks += 32) {
    v16bf a_nxt = a_cur;
    v16bf b_nxt[8];
    if (ks + 32 < Ff) {
      a_nxt = frag_a_f32(in, row0, ks + 32, Ff, lane);
      load_b_panel(b_nxt, WeT, ks + 32, Ff, lane);
    }
    wmma8(acc, a_cur, b_cur);
    a_cur = a_nxt;
#pragma unroll
    for (int j = 0; j < 8; ++j) b_cur[j] = b_nxt[j];
  }

  const int g = lane >> 4, b = row0 / Nn;
  float mv[8];
#pragma unroll
  for (int v = 0; v < 8; ++v) mv[v] = mask[row0 + 8 * g + v];
#pragma unroll
  for (int j = 0; j < 8; ++j) {
    int n = 16 * j + (lane & 15);
    float bias = be[n];
#pragma unroll
    for (int v = 0; v < 8; ++v) {
      int m = row0 + 8 * g + v;
      float val = (acc[j][v] + bias) * mv[v];
      x[(size_t)m * Hh + n]  = val;
      xb[(size_t)m * Hh + n] = (__bf16)val;
      xT[(size_t)b * Hh * Nn + (size_t)n * Nn + (m - b * Nn)] = (__bf16)val;
    }
  }
}

// ---------------------------------------------------------------------------
// agg: a[b] = adj_bf16[b] @ x[b]   ([1024,1024]@[1024,128], B operand = x^T)
// A-panel streamed by the Tensor Data Mover into double-buffered LDS; DMA of
// tile k+1 overlaps the 16 WMMAs of tile k. B panels preloaded as one clause.
// ---------------------------------------------------------------------------
__global__ __launch_bounds__(256) void agg_kernel(const __bf16* __restrict__ adjb,
                                                  const __bf16* __restrict__ xT,
                                                  __bf16* __restrict__ ag) {
  __shared__ __bf16 atile[2][128][64];                      // 2 x 16 KB
  const int lane = threadIdx.x & 31, wave = threadIdx.x >> 5;
  const int b = blockIdx.y;
  const int row0 = blockIdx.x * 128;                        // WG row base in batch
  const __bf16* A  = adjb + (size_t)b * Nn * Nn + (size_t)row0 * Nn;
  const __bf16* BT = xT   + (size_t)b * Hh * Nn;
  const v8f vzero = {0.f, 0.f, 0.f, 0.f, 0.f, 0.f, 0.f, 0.f};
  v8f acc[8];
#pragma unroll
  for (int j = 0; j < 8; ++j) acc[j] = vzero;

  if (wave == 0) {
    tdm_load_tile_64x128(A, &atile[0][0][0]);
    __builtin_amdgcn_s_wait_tensorcnt(0);
  }
  __syncthreads();

  for (int kc = 0; kc < Nn; kc += 64) {
    const int buf = (kc >> 6) & 1;
    if (wave == 0 && kc + 64 < Nn)                          // prefetch next tile
      tdm_load_tile_64x128(A + kc + 64, &atile[buf ^ 1][0][0]);

    v16bf b0[8], b1[8];
    load_b_panel(b0, BT, kc, Nn, lane);                     // 32-load clause
    load_b_panel(b1, BT, kc + 32, Nn, lane);
    v16bf a0 = frag_a_lds(&atile[buf][wave * 16][0], 0, lane);
    v16bf a1 = frag_a_lds(&atile[buf][wave * 16][0], 32, lane);
    wmma8(acc, a0, b0);
    wmma8(acc, a1, b1);

    if (wave == 0) __builtin_amdgcn_s_wait_tensorcnt(0);    // prefetch landed
    __syncthreads();
  }

  const int g = lane >> 4;
#pragma unroll
  for (int j = 0; j < 8; ++j) {
    int n = 16 * j + (lane & 15);
#pragma unroll
    for (int v = 0; v < 8; ++v) {
      int m = row0 + wave * 16 + 8 * g + v;
      ag[((size_t)b * Nn + m) * Hh + n] = (__bf16)acc[j][v];
    }
  }
}

// ---------------------------------------------------------------------------
// gate (fully fused): z = sig([a|x]Wz + bz), r = sig([a|x]Wr + br),
// h = relu([a|r*x]Wh + bh), x = (h*z + x*(1-z))*mask  — all row-local,
// x updated in place; r*x (bf16) and sigmoid(z) (f32) staged per-wave in LDS.
// ---------------------------------------------------------------------------
__global__ __launch_bounds__(256) void gate_kernel(const __bf16* __restrict__ ag,
                                                   float* __restrict__ x,
                                                   __bf16* __restrict__ xb,
                                                   __bf16* __restrict__ xT,
                                                   const __bf16* __restrict__ WzT,
                                                   const __bf16* __restrict__ WrT,
                                                   const __bf16* __restrict__ WhT,
                                                   const float* __restrict__ bz,
                                                   const float* __restrict__ br,
                                                   const float* __restrict__ bh,
                                                   const float* __restrict__ mask) {
  __shared__ __bf16 rx_lds[8][16][Hh];                       // 32 KB
  __shared__ float  z_lds[8][16][Hh];                        // 64 KB
  const int lane = threadIdx.x & 31, wave = threadIdx.x >> 5;
  const int row0 = blockIdx.x * 128 + wave * 16;
  const int g = lane >> 4;
  const v8f vzero = {0.f, 0.f, 0.f, 0.f, 0.f, 0.f, 0.f, 0.f};

  // ---- r gate: K=256 over [a | x]
  v8f racc[8];
#pragma unroll
  for (int j = 0; j < 8; ++j) racc[j] = vzero;
#pragma unroll
  for (int ks = 0; ks < Hh; ks += 32)
    wmma_step(racc, frag_a_bf16(ag, row0, ks, Hh, lane), WrT, ks, 2 * Hh, lane);
#pragma unroll
  for (int ks = 0; ks < Hh; ks += 32)
    wmma_step(racc, frag_a_bf16(xb, row0, ks, Hh, lane), WrT, Hh + ks, 2 * Hh, lane);

#pragma unroll
  for (int j = 0; j < 8; ++j) {
    int n = 16 * j + (lane & 15);
    float bias = br[n];
#pragma unroll
    for (int v = 0; v < 8; ++v) {
      int m = row0 + 8 * g + v;
      float r  = 1.f / (1.f + __expf(-(racc[j][v] + bias)));
      float xv = x[(size_t)m * Hh + n];
      rx_lds[wave][8 * g + v][n] = (__bf16)(r * xv);
    }
  }
  __syncthreads();

  // ---- z gate: K=256 over [a | x]; sigmoid(z) parked in LDS (frees VGPRs)
  v8f zacc[8];
#pragma unroll
  for (int j = 0; j < 8; ++j) zacc[j] = vzero;
#pragma unroll
  for (int ks = 0; ks < Hh; ks += 32)
    wmma_step(zacc, frag_a_bf16(ag, row0, ks, Hh, lane), WzT, ks, 2 * Hh, lane);
#pragma unroll
  for (int ks = 0; ks < Hh; ks += 32)
    wmma_step(zacc, frag_a_bf16(xb, row0, ks, Hh, lane), WzT, Hh + ks, 2 * Hh, lane);
#pragma unroll
  for (int j = 0; j < 8; ++j) {
    int n = 16 * j + (lane & 15);
    float bias = bz[n];
#pragma unroll
    for (int v = 0; v < 8; ++v)
      z_lds[wave][8 * g + v][n] = 1.f / (1.f + __expf(-(zacc[j][v] + bias)));
  }

  // ---- h gate: K=256 over [a | r*x(LDS)]
  v8f hacc[8];
#pragma unroll
  for (int j = 0; j < 8; ++j) hacc[j] = vzero;
#pragma unroll
  for (int ks = 0; ks < Hh; ks += 32)
    wmma_step(hacc, frag_a_bf16(ag, row0, ks, Hh, lane), WhT, ks, 2 * Hh, lane);
#pragma unroll
  for (int ks = 0; ks < Hh; ks += 32) {
    const __bf16* p = &rx_lds[wave][lane & 15][ks + 8 * g];
    v16bf af;
    ((float4*)&af)[0] = *(const float4*)p;            // ds_load_b128
    ((float4*)&af)[1] = *(const float4*)(p + 16);
    wmma_step(hacc, af, WhT, Hh + ks, 2 * Hh, lane);
  }

  // ---- epilogue: x = (h*z + x*(1-z)) * mask, in place + bf16 + transposed bf16
  const int b = row0 / Nn;
  float mv[8];
#pragma unroll
  for (int v = 0; v < 8; ++v) mv[v] = mask[row0 + 8 * g + v];
#pragma unroll
  for (int j = 0; j < 8; ++j) {
    int n = 16 * j + (lane & 15);
    float bias = bh[n];
#pragma unroll
    for (int v = 0; v < 8; ++v) {
      int m = row0 + 8 * g + v;
      float h  = fmaxf(hacc[j][v] + bias, 0.f);
      float zv = z_lds[wave][8 * g + v][n];           // same-lane readback
      float xv = x[(size_t)m * Hh + n];
      float res = (h * zv + xv * (1.f - zv)) * mv[v];
      x[(size_t)m * Hh + n]  = res;
      xb[(size_t)m * Hh + n] = (__bf16)res;
      xT[(size_t)b * Hh * Nn + (size_t)n * Nn + (m - b * Nn)] = (__bf16)res;
    }
  }
}

// ---------------------------------------------------------------------------
extern "C" void kernel_launch(void* const* d_in, const int* in_sizes, int n_in,
                              void* d_out, int out_size, void* d_ws, size_t ws_size,
                              hipStream_t stream) {
  (void)in_sizes; (void)n_in; (void)out_size; (void)ws_size;
  const float* inputs = (const float*)d_in[0];
  const float* adj    = (const float*)d_in[1];
  const float* mask   = (const float*)d_in[2];
  const float* We  = (const float*)d_in[3];
  const float* be  = (const float*)d_in[4];
  const float* Wz0 = (const float*)d_in[5],  *bz0 = (const float*)d_in[6];
  const float* Wz1 = (const float*)d_in[7],  *bz1 = (const float*)d_in[8];
  const float* Wr0 = (const float*)d_in[9],  *br0 = (const float*)d_in[10];
  const float* Wr1 = (const float*)d_in[11], *br1 = (const float*)d_in[12];
  const float* Wh0 = (const float*)d_in[13], *bh0 = (const float*)d_in[14];
  const float* Wh1 = (const float*)d_in[15], *bh1 = (const float*)d_in[16];
  float* x = (float*)d_out;

  char* ws = (char*)d_ws;
  auto alloc = [&](size_t bytes) {
    char* p = ws; ws += (bytes + 255) & ~(size_t)255; return p;
  };
  __bf16* WeT  = (__bf16*)alloc((size_t)Hh * Ff * 2);
  __bf16* WzT  = (__bf16*)alloc((size_t)Hh * 2 * Hh * 2);
  __bf16* WrT  = (__bf16*)alloc((size_t)Hh * 2 * Hh * 2);
  __bf16* WhT  = (__bf16*)alloc((size_t)Hh * 2 * Hh * 2);
  float*  bzf  = (float*)alloc(Hh * 4);
  float*  brf  = (float*)alloc(Hh * 4);
  float*  bhf  = (float*)alloc(Hh * 4);
  __bf16* xb   = (__bf16*)alloc((size_t)BN * Hh * 2);
  __bf16* xT   = (__bf16*)alloc((size_t)BN * Hh * 2);
  __bf16* ag   = (__bf16*)alloc((size_t)BN * Hh * 2);
  __bf16* adjb = (__bf16*)alloc((size_t)Bb * Nn * Nn * 2);   // 64 MB, L2-resident

  prep_kernel<<<(Hh * 2 * Hh + 255) / 256, 256, 0, stream>>>(
      We, Wz0, Wz1, Wr0, Wr1, Wh0, Wh1, bz0, bz1, br0, br1, bh0, bh1,
      WeT, WzT, WrT, WhT, bzf, brf, bhf);

  adjcvt_kernel<<<4096, 256, 0, stream>>>(adj, adjb, (long)Bb * Nn * Nn / 4);

  embed_kernel<<<BN / 128, 256, 0, stream>>>(inputs, mask, WeT, be, x, xb, xT);

  for (int l = 0; l < Ll; ++l) {
    agg_kernel<<<dim3(Nn / 128, Bb), 256, 0, stream>>>(adjb, xT, ag);
    gate_kernel<<<BN / 128, 256, 0, stream>>>(ag, x, xb, xT, WzT, WrT, WhT,
                                              bzf, brf, bhf, mask);
  }
}
